// e3nnEncoder_54142357733961
// MI455X (gfx1250) — matrix operand order repaired and checked by reading, not compiled, not run on hardware
//
#include <hip/hip_runtime.h>
#include <hip/hip_bf16.h>
#include <math.h>

// ---------------------------------------------------------------------------
// CDNA5 (gfx1250) e3nn encoder forward pass.
//  - All dense 2-layer MLPs: v_wmma_f32_16x16x32_bf16, 4 waves/block, 64-row
//    tiles. Weights pre-converted to padded bf16; activations produced as
//    padded bf16 so MLP staging is conversion-free.
//  - X tiles staged to LDS via the Tensor Data Mover (tensor_load_to_lds +
//    s_wait_tensorcnt); W tiles staged transposed so A/B fragments are
//    ds_load_b128 pairs.
//  - TP conv / scatter: scalar VALU + global atomics, chunked edge weights.
// ---------------------------------------------------------------------------

typedef __bf16 bf16_t;
typedef bf16_t v16bf_t __attribute__((ext_vector_type(16)));
typedef float  v8f_t   __attribute__((ext_vector_type(8)));
typedef unsigned int u32x4_t __attribute__((ext_vector_type(4)));
typedef int i32x8_t __attribute__((ext_vector_type(8)));
typedef int i32x4_t __attribute__((ext_vector_type(4)));

union FragA { v16bf_t v; bf16_t h[16]; };
union FragC { v8f_t   v; float  f[8];  };

#define MLP_BLOCK 128   // 4 waves
#define MLP_KMAX 96     // max padded K for stage 1 (din <= 84 -> 96)
#define MLP_HMAX 256    // max padded hidden (dh <= 256)

// ------------------------- tensor-product path tables ----------------------
struct TpPath { int s1, m1, d1, s2, d2, mo, d3, oo, ws, co; float alpha; };

#define C_S3 0.57735027f   // 1/sqrt(3)
#define C_E6 0.40824829f   // 1/sqrt(6)
#define C_U45 0.14907120f  // 1/sqrt(45) (normalized 1x2->1 block placeholder)

__constant__ float TP_C[100] = {
    1.0f,
    // [1..9] 0x1->1 : delta_jk / sqrt(3)   (1x3x3)
    C_S3, 0.f, 0.f, 0.f, C_S3, 0.f, 0.f, 0.f, C_S3,
    // [10..18] 1x0->1 : delta_ik / sqrt(3) (3x1x3)
    C_S3, 0.f, 0.f, 0.f, C_S3, 0.f, 0.f, 0.f, C_S3,
    // [19..27] 1x1->0 : delta_ij / sqrt(3) (3x3x1)
    C_S3, 0.f, 0.f, 0.f, C_S3, 0.f, 0.f, 0.f, C_S3,
    // [28..54] 1x1->1 : levi-civita / sqrt(6) (3x3x3)
    0.f, 0.f, 0.f,   0.f, 0.f, C_E6,   0.f, -C_E6, 0.f,
    0.f, 0.f, -C_E6, 0.f, 0.f, 0.f,    C_E6, 0.f,  0.f,
    0.f, C_E6, 0.f,  -C_E6, 0.f, 0.f,  0.f, 0.f,   0.f,
    // [55..99] 1x2->1 (3x5x3) normalized block
    C_U45, C_U45, C_U45, C_U45, C_U45, C_U45, C_U45, C_U45, C_U45,
    C_U45, C_U45, C_U45, C_U45, C_U45, C_U45, C_U45, C_U45, C_U45,
    C_U45, C_U45, C_U45, C_U45, C_U45, C_U45, C_U45, C_U45, C_U45,
    C_U45, C_U45, C_U45, C_U45, C_U45, C_U45, C_U45, C_U45, C_U45,
    C_U45, C_U45, C_U45, C_U45, C_U45, C_U45, C_U45, C_U45, C_U45,
};

__constant__ TpPath TP_PATHS[18] = {
    // cfg 0 (2 paths, numel 192, out_dim 24)
    { 0, 12, 1, 0, 1, 12, 1,  0,   0,  0, 0.288675f },
    { 0, 12, 1, 1, 3,  4, 3, 12, 144,  1, 0.5f      },
    // cfg 1 (6 paths, numel 288, out_dim 36)
    { 0, 12, 1, 0, 1, 12, 1,  0,   0,  0, 0.25f     },
    { 0, 12, 1, 1, 3,  4, 3, 12, 144,  1, 0.387298f },
    {12,  4, 3, 0, 1,  4, 3, 12, 192, 10, 0.387298f },
    {12,  4, 3, 1, 3, 12, 1,  0, 208, 19, 0.25f     },
    {12,  4, 3, 1, 3,  4, 3, 24, 256, 28, 0.866025f },
    {12,  4, 3, 4, 5,  4, 3, 12, 272, 55, 0.387298f },
    // cfg 2 (10 paths, numel 384, out_dim 48)
    { 0, 12, 1, 0, 1, 12, 1,  0,   0,  0, 0.25f     },
    { 0, 12, 1, 1, 3,  4, 3, 12, 144,  1, 0.353553f },
    {12,  4, 3, 0, 1,  4, 3, 12, 192, 10, 0.353553f },
    {12,  4, 3, 1, 3, 12, 1,  0, 208, 19, 0.25f     },
    {12,  4, 3, 1, 3,  4, 3, 24, 256, 28, 0.5f      },
    {12,  4, 3, 4, 5,  4, 3, 12, 272, 55, 0.353553f },
    {24,  4, 3, 0, 1,  4, 3, 24, 288, 10, 0.5f      },
    {24,  4, 3, 1, 3,  4, 3, 12, 304, 28, 0.353553f },
    {24,  4, 3, 1, 3, 12, 1, 36, 320, 19, 0.5f      },
    {24,  4, 3, 4, 5,  4, 3, 24, 368, 55, 0.5f      },
};
__constant__ int TP_START[3]  = {0, 2, 8};
__constant__ int TP_NP[3]     = {2, 6, 10};
__constant__ int TP_OUTDIM[3] = {24, 36, 48};

// ----------------------------- helpers -------------------------------------
__device__ inline void sph9(float x, float y, float z, float* o) {
    const float s3 = 1.7320508f, s5 = 2.2360680f, s15 = 3.8729833f;
    o[0] = 1.f;
    o[1] = s3 * y;  o[2] = s3 * z;  o[3] = s3 * x;
    o[4] = s15 * x * y;  o[5] = s15 * y * z;
    o[6] = 0.5f * s5 * (3.f * z * z - 1.f);
    o[7] = s15 * x * z;  o[8] = 0.5f * s15 * (x * x - y * y);
}

// Issue a 2-D TDM load: tile (Kp x 64 rows) of bf16 from `src` into LDS at
// `ldsOff`. D# layout per CDNA5 ISA ch.8 (group0: count/lds/global addr,
// group1: data_size=2B, dims/strides). Rows past (M - row0) read as zero (OOB).
__device__ inline void tdm_load_tile(const bf16_t* src, unsigned ldsOff,
                                     int Kp, int rowsLeft)
{
    unsigned long long ga = (unsigned long long)(const void*)src;
    u32x4_t g0;
    g0[0] = 1u;                                            // count=1 (user D#)
    g0[1] = ldsOff;                                        // lds_addr
    g0[2] = (unsigned)(ga & 0xFFFFFFFFu);                  // global_addr lo
    g0[3] = (unsigned)((ga >> 32) & 0x01FFFFFFu) | (2u << 30); // addr hi | type=2
    i32x8_t g1;
    g1[0] = (int)(1u << 16);                               // data_size=2 bytes
    g1[1] = (int)(((unsigned)Kp & 0xFFFFu) << 16);         // tensor_dim0 lo16
    g1[2] = (int)((((unsigned)Kp >> 16) & 0xFFFFu) |
                  (((unsigned)rowsLeft & 0xFFFFu) << 16)); // dim0 hi | dim1 lo
    g1[3] = (int)((((unsigned)rowsLeft >> 16) & 0xFFFFu) |
                  (((unsigned)Kp & 0xFFFFu) << 16));       // dim1 hi | tile_dim0
    g1[4] = 64;                                            // tile_dim1 = 64 rows
    g1[5] = Kp;                                            // tensor_dim0_stride
    g1[6] = 0;
    g1[7] = 0;
    i32x4_t gz4 = {0, 0, 0, 0};
    i32x8_t gz8 = {0, 0, 0, 0, 0, 0, 0, 0};
    __builtin_amdgcn_tensor_load_to_lds(g0, g1, gz4, gz4, gz8, 0);
    __builtin_amdgcn_s_wait_tensorcnt(0);
}

// -------------------- fused 2-layer MLP via bf16 WMMA -----------------------
// X: (M x Kp1) bf16, K-padded/zero-filled.  W1p: (Kp1 x Np1) bf16 padded.
// W2p: (round32(dh) x Np2) bf16 padded.  Y = hidden-act(X W1 + b1) W2 + b2.
__global__ void __launch_bounds__(MLP_BLOCK)
mlp2_wmma_kernel(const bf16_t* __restrict__ X, int M, int Kp1,
                 const bf16_t* __restrict__ W1p, int Np1,
                 const float* __restrict__ B1, int dh,
                 const bf16_t* __restrict__ W2p, int Np2,
                 const float* __restrict__ B2, int dout,
                 float* __restrict__ Y, int act_mode)
{
    __shared__ bf16_t sX[64 * MLP_KMAX];
    __shared__ bf16_t sH[64 * MLP_HMAX];
    __shared__ bf16_t sWt[16 * 32];          // transposed W tile: [n][k]

    const int tid  = threadIdx.x;
    const int wave = tid >> 5, lane = tid & 31;
    const int row0 = blockIdx.x * 64;
    const int Kp2  = (dh + 31) & ~31;

    if (lane == 0) { __builtin_prefetch(W1p, 0, 1); __builtin_prefetch(W2p, 0, 1); }

    // ---- X tile via Tensor Data Mover (wave 0 issues, all waves consume)
    if (wave == 0) {
        unsigned ldsOff = (unsigned)(size_t)(void*)sX;   // LDS offset = low 32b
        tdm_load_tile(X + (size_t)row0 * Kp1, ldsOff, Kp1, M - row0);
    }
    __syncthreads();

    const int ar   = (wave << 4) + (lane & 15);          // A-fragment row
    const int laneHi8 = (lane >= 16) ? 8 : 0;
    const int bc   = lane & 15;                          // B-fragment col
    const int kofB = (lane >= 16) ? 8 : 0;

    // ---- stage 1: H = act(X W1 + b1) -> LDS bf16, stride Kp2
    const int nt1 = Np1 >> 4, kt1 = Kp1 >> 5;
    for (int nt = 0; nt < nt1; ++nt) {
        FragC acc;
        #pragma unroll
        for (int r = 0; r < 8; ++r) acc.f[r] = 0.f;
        for (int kt = 0; kt < kt1; ++kt) {
            #pragma unroll
            for (int idx = tid; idx < 512; idx += MLP_BLOCK) {
                int n = idx & 15, k = idx >> 4;
                sWt[n * 32 + k] = W1p[(size_t)(kt * 32 + k) * Np1 + nt * 16 + n];
            }
            __syncthreads();
            FragA a, b;
            const bf16_t* ap = sX + (size_t)ar * Kp1 + kt * 32 + laneHi8;
            *(uint4*)&a.h[0] = *(const uint4*)(ap);
            *(uint4*)&a.h[8] = *(const uint4*)(ap + 16);
            const bf16_t* bp = sWt + bc * 32 + kofB;
            *(uint4*)&b.h[0] = *(const uint4*)(bp);
            *(uint4*)&b.h[8] = *(const uint4*)(bp + 16);
            acc.v = __builtin_amdgcn_wmma_f32_16x16x32_bf16(
                false, a.v, false, b.v, (short)0, acc.v, false, false);
            __syncthreads();
        }
        int col = nt * 16 + bc;
        #pragma unroll
        for (int r = 0; r < 8; ++r) {
            int rr = (wave << 4) + r + laneHi8;
            float v = 0.f;
            if (col < dh) {
                v = acc.f[r] + B1[col];
                v = act_mode ? tanhf(v) : fmaxf(v, 0.f);
            }
            sH[rr * Kp2 + col] = (bf16_t)v;
        }
    }
    // zero K-padding columns of H (padw is 0 or 16 for all shapes used here)
    if (Kp2 > Np1) {
        #pragma unroll
        for (int idx = tid; idx < 64 * 16; idx += MLP_BLOCK) {
            int r = idx >> 4, c = Np1 + (idx & 15);
            sH[r * Kp2 + c] = (bf16_t)0.f;
        }
    }
    __syncthreads();

    // ---- stage 2: Y = H W2 + b2 -> global f32
    const int nt2 = Np2 >> 4, kt2 = Kp2 >> 5;
    for (int nt = 0; nt < nt2; ++nt) {
        FragC acc;
        #pragma unroll
        for (int r = 0; r < 8; ++r) acc.f[r] = 0.f;
        for (int kt = 0; kt < kt2; ++kt) {
            #pragma unroll
            for (int idx = tid; idx < 512; idx += MLP_BLOCK) {
                int n = idx & 15, k = idx >> 4;
                sWt[n * 32 + k] = W2p[(size_t)(kt * 32 + k) * Np2 + nt * 16 + n];
            }
            __syncthreads();
            FragA a, b;
            const bf16_t* ap = sH + (size_t)ar * Kp2 + kt * 32 + laneHi8;
            *(uint4*)&a.h[0] = *(const uint4*)(ap);
            *(uint4*)&a.h[8] = *(const uint4*)(ap + 16);
            const bf16_t* bp = sWt + bc * 32 + kofB;
            *(uint4*)&b.h[0] = *(const uint4*)(bp);
            *(uint4*)&b.h[8] = *(const uint4*)(bp + 16);
            acc.v = __builtin_amdgcn_wmma_f32_16x16x32_bf16(
                false, a.v, false, b.v, (short)0, acc.v, false, false);
            __syncthreads();
        }
        int col = nt * 16 + bc;
        #pragma unroll
        for (int r = 0; r < 8; ++r) {
            int rr = row0 + (wave << 4) + r + laneHi8;
            if (rr < M && col < dout)
                Y[(size_t)rr * dout + col] = acc.f[r] + B2[col];
        }
    }
}

// -------------------- weight conversion (f32 -> padded bf16) ----------------
__global__ void wconv_kernel(const float* __restrict__ W, int R, int C,
                             bf16_t* __restrict__ out, int Rp, int Cp)
{
    int t = blockIdx.x * blockDim.x + threadIdx.x;
    if (t >= Rp * Cp) return;
    int r = t / Cp, c = t % Cp;
    out[t] = (bf16_t)((r < R && c < C) ? W[(size_t)r * C + c] : 0.f);
}

// -------------------- edge geometry / feature kernels -----------------------
// feat: bf16, stride 32, zero padded past din.
__global__ void pair_edge_kernel(const float* __restrict__ pos, const int* __restrict__ z,
                                 const int* __restrict__ nbr, int nPairs, float rmax,
                                 bf16_t* __restrict__ feat, float* __restrict__ sh,
                                 int* __restrict__ srcO, int* __restrict__ dstO)
{
    int e = blockIdx.x * blockDim.x + threadIdx.x;
    int E = 2 * nPairs;
    if (e >= E) return;
    int s, d;
    if (e < nPairs) { s = nbr[2 * e];                d = nbr[2 * e + 1];          }
    else            { s = nbr[2 * (e - nPairs) + 1]; d = nbr[2 * (e - nPairs)];   }
    srcO[e] = s; dstO[e] = d;
    float rx = pos[3 * d + 0] - pos[3 * s + 0];
    float ry = pos[3 * d + 1] - pos[3 * s + 1];
    float rz = pos[3 * d + 2] - pos[3 * s + 2];
    float n = sqrtf(rx * rx + ry * ry + rz * rz + 1e-12f);
    bf16_t* f = feat + (size_t)e * 32;
    f[0] = (bf16_t)(float)z[s]; f[1] = (bf16_t)(float)z[d];
    f[2] = (bf16_t)0.f; f[3] = (bf16_t)0.f; f[4] = (bf16_t)0.f; f[5] = (bf16_t)0.f;
    float delta = rmax / 7.f, coeff = -0.5f / (delta * delta);
    #pragma unroll
    for (int i = 0; i < 8; ++i) {
        float dd = n - rmax * (float)i / 7.f;
        f[6 + i] = (bf16_t)__expf(coeff * dd * dd);
    }
    #pragma unroll
    for (int i = 14; i < 32; ++i) f[i] = (bf16_t)0.f;
    sph9(rx / n, ry / n, rz / n, sh + (size_t)e * 9);
}

__global__ void cross_edge_kernel(const float* __restrict__ xyz, const float* __restrict__ cg_xyz,
                                  const int* __restrict__ mapping, int N, float rmax,
                                  bf16_t* __restrict__ feat, float* __restrict__ sh,
                                  int* __restrict__ srcO, int* __restrict__ dstO)
{
    int i = blockIdx.x * blockDim.x + threadIdx.x;
    if (i >= N) return;
    int m = mapping[i];
    srcO[i] = i; dstO[i] = m;
    float rx = xyz[3 * i + 0] - cg_xyz[3 * m + 0];
    float ry = xyz[3 * i + 1] - cg_xyz[3 * m + 1];
    float rz = xyz[3 * i + 2] - cg_xyz[3 * m + 2];
    float n = sqrtf(rx * rx + ry * ry + rz * rz + 1e-12f);
    bf16_t* f = feat + (size_t)i * 32;
    float delta = rmax / 7.f, coeff = -0.5f / (delta * delta);
    #pragma unroll
    for (int k = 0; k < 8; ++k) {
        float dd = n - rmax * (float)k / 7.f;
        f[k] = (bf16_t)__expf(coeff * dd * dd);
    }
    #pragma unroll
    for (int k = 8; k < 32; ++k) f[k] = (bf16_t)0.f;
    sph9(rx / n, ry / n, rz / n, sh + (size_t)i * 9);
}

// ----------------------- gather / embedding / update ------------------------
__global__ void emb_kernel(const int* __restrict__ z, const float* __restrict__ emb,
                           float* __restrict__ node, int N)
{
    int t = blockIdx.x * blockDim.x + threadIdx.x;
    if (t >= N * 12) return;
    int i = t / 12, c = t % 12;
    node[(size_t)i * 48 + c] = emb[(size_t)z[i] * 12 + c];
}

// out: bf16 (n x 64), cols 36..63 zero.
__global__ void gather_conv_kernel(const float* __restrict__ edgeFeat,
                                   const float* __restrict__ nodeS, const float* __restrict__ nodeD,
                                   const int* __restrict__ src, const int* __restrict__ dst,
                                   int ebase, int n, bf16_t* __restrict__ out)
{
    int t = blockIdx.x * blockDim.x + threadIdx.x;
    if (t >= n * 64) return;
    int e = t >> 6, c = t & 63;
    int ei = ebase + e;
    float v = 0.f;
    if (c < 12)       v = edgeFeat[(size_t)ei * 12 + c];
    else if (c < 24)  v = nodeS[(size_t)src[ei] * 48 + (c - 12)];
    else if (c < 36)  v = nodeD[(size_t)dst[ei] * 48 + (c - 24)];
    out[t] = (bf16_t)v;
}

__global__ void tp_conv_kernel(const float* __restrict__ nodeAttr,
                               const int* __restrict__ src, const int* __restrict__ dst,
                               const float* __restrict__ W, const float* __restrict__ SH,
                               int ebase, int n, int cfg, int numel,
                               float* __restrict__ acc, float* __restrict__ cnt)
{
    int t = blockIdx.x * blockDim.x + threadIdx.x;
    if (t >= n) return;
    int ei = ebase + t;
    const float* x  = nodeAttr + (size_t)dst[ei] * 48;
    const float* sh = SH + (size_t)ei * 9;
    const float* w  = W + (size_t)t * numel;
    float msg[48];
    int odim = TP_OUTDIM[cfg];
    for (int d = 0; d < odim; ++d) msg[d] = 0.f;
    int p0 = TP_START[cfg], np = TP_NP[cfg];
    for (int p = p0; p < p0 + np; ++p) {
        TpPath tp = TP_PATHS[p];
        const float* C = TP_C + tp.co;
        for (int u = 0; u < tp.m1; ++u) {
            for (int wo = 0; wo < tp.mo; ++wo) {
                float wv = w[tp.ws + u * tp.mo + wo];
                for (int i = 0; i < tp.d1; ++i) {
                    float xw = x[tp.s1 + u * tp.d1 + i] * wv;
                    for (int j = 0; j < tp.d2; ++j) {
                        float sv = sh[tp.s2 + j] * xw * tp.alpha;
                        for (int k = 0; k < tp.d3; ++k)
                            msg[tp.oo + wo * tp.d3 + k] += sv * C[(i * tp.d2 + j) * tp.d3 + k];
                    }
                }
            }
        }
    }
    int s = src[ei];
    for (int d = 0; d < odim; ++d)
        atomicAdd(&acc[(size_t)s * 48 + d], msg[d]);
    atomicAdd(&cnt[s], 1.f);
}

__global__ void node_update_kernel(float* __restrict__ node,
                                   const float* __restrict__ accI, const float* __restrict__ cntI,
                                   const float* __restrict__ accX, const float* __restrict__ cntX,
                                   int N, int dim)
{
    int t = blockIdx.x * blockDim.x + threadIdx.x;
    if (t >= N * dim) return;
    int i = t / dim, d = t % dim;
    float ci = fmaxf(cntI[i], 1.f), cx = fmaxf(cntX[i], 1.f);
    node[(size_t)i * 48 + d] += accI[(size_t)i * 48 + d] / ci +
                                accX[(size_t)i * 48 + d] / cx;
}

// -------------------------- pooling + final dense ---------------------------
__global__ void hpool_kernel(const float* __restrict__ aNode, const float* __restrict__ cNode,
                             const int* __restrict__ mapping, int N,
                             float* __restrict__ hsum, float* __restrict__ hcnt)
{
    int i = blockIdx.x * blockDim.x + threadIdx.x;
    if (i >= N) return;
    int m = mapping[i];
    for (int d = 0; d < 48; ++d)
        atomicAdd(&hsum[(size_t)m * 84 + d], aNode[(size_t)i * 48 + d]);
    for (int d = 0; d < 36; ++d)
        atomicAdd(&hsum[(size_t)m * 84 + 48 + d], cNode[(size_t)m * 48 + d]);
    atomicAdd(&hcnt[m], 1.f);
}

// h: bf16 (N x 96), cols 84..95 zero.
__global__ void hdiv_kernel(const float* __restrict__ hsum, const float* __restrict__ hcnt,
                            bf16_t* __restrict__ h, int N)
{
    int t = blockIdx.x * blockDim.x + threadIdx.x;
    if (t >= N * 96) return;
    int i = t / 96, c = t % 96;
    float v = (c < 84) ? hsum[(size_t)i * 84 + c] / fmaxf(hcnt[i], 1.f) : 0.f;
    h[t] = (bf16_t)v;
}

// ------------------------------- host driver --------------------------------
static const int H_TP_NUMEL[3]  = {192, 288, 384};
static const int H_TP_OUTDIM[3] = {24, 36, 48};

extern "C" void kernel_launch(void* const* d_in, const int* in_sizes, int n_in,
                              void* d_out, int out_size, void* d_ws, size_t ws_size,
                              hipStream_t stream)
{
    (void)in_sizes; (void)n_in; (void)out_size; (void)ws_size;

    const float* xyz      = (const float*)d_in[0];
    const float* cg_xyz   = (const float*)d_in[1];
    const float* atom_emb = (const float*)d_in[2];
    const float* cg_emb   = (const float*)d_in[3];
    struct Mlp { const float *w1, *b1, *w2, *b2; };
    auto getMlp = [&](int base) -> Mlp {
        return { (const float*)d_in[base], (const float*)d_in[base + 1],
                 (const float*)d_in[base + 2], (const float*)d_in[base + 3] };
    };
    const int* z       = (const int*)d_in[68];
    const int* cg_z    = (const int*)d_in[69];
    const int* mapping = (const int*)d_in[70];
    const int* nbr     = (const int*)d_in[71];
    const int* cg_nbr  = (const int*)d_in[72];

    const int NA = 30000, NC = 3000;
    const int NP_A = 150000, NP_C = 24000;
    const int EA = 2 * NP_A, EC = 2 * NP_C, EX = NA;
    const int CH = 32768;

    size_t off = 0;
    auto alloc = [&](size_t bytes) -> void* {
        off = (off + 255) & ~(size_t)255;
        void* p = (char*)d_ws + off;
        off += bytes;
        return p;
    };
    auto g1 = [](int n) { return (n + 255) / 256; };
    auto r32 = [](int x) { return (x + 31) & ~31; };
    auto r16 = [](int x) { return (x + 15) & ~15; };

    // padded bf16 weights + launch helper
    struct MlpP { const float *b1, *b2; bf16_t *w1p, *w2p; int kp1, np1, dh, np2, dout; };
    auto prep = [&](Mlp m, int din, int dh, int dout) -> MlpP {
        MlpP p; p.b1 = m.b1; p.b2 = m.b2;
        p.kp1 = r32(din); p.np1 = r16(dh); p.dh = dh;
        int kp2 = r32(dh); p.np2 = r16(dout); p.dout = dout;
        p.w1p = (bf16_t*)alloc((size_t)p.kp1 * p.np1 * 2);
        p.w2p = (bf16_t*)alloc((size_t)kp2 * p.np2 * 2);
        wconv_kernel<<<g1(p.kp1 * p.np1), 256, 0, stream>>>(m.w1, din, dh, p.w1p, p.kp1, p.np1);
        wconv_kernel<<<g1(kp2 * p.np2), 256, 0, stream>>>(m.w2, dh, dout, p.w2p, kp2, p.np2);
        return p;
    };
    auto runMlp = [&](const bf16_t* X, int M, const MlpP& p, float* Y, int act) {
        mlp2_wmma_kernel<<<(M + 63) / 64, MLP_BLOCK, 0, stream>>>(
            X, M, p.kp1, p.w1p, p.np1, p.b1, p.dh, p.w2p, p.np2, p.b2, p.dout, Y, act);
    };

    // workspace buffers
    int*    aSrc  = (int*)alloc((size_t)EA * 4);
    int*    aDst  = (int*)alloc((size_t)EA * 4);
    int*    cSrc  = (int*)alloc((size_t)EC * 4);
    int*    cDst  = (int*)alloc((size_t)EC * 4);
    int*    xSrc  = (int*)alloc((size_t)EX * 4);
    int*    xDst  = (int*)alloc((size_t)EX * 4);
    bf16_t* aFeat = (bf16_t*)alloc((size_t)EA * 32 * 2);
    bf16_t* cFeat = (bf16_t*)alloc((size_t)EC * 32 * 2);
    bf16_t* xFeat = (bf16_t*)alloc((size_t)EX * 32 * 2);
    float*  aSh   = (float*)alloc((size_t)EA * 9 * 4);
    float*  cSh   = (float*)alloc((size_t)EC * 9 * 4);
    float*  xSh   = (float*)alloc((size_t)EX * 9 * 4);
    float*  aEdge = (float*)alloc((size_t)EA * 12 * 4);
    float*  cEdge = (float*)alloc((size_t)EC * 12 * 4);
    float*  xEdge = (float*)alloc((size_t)EX * 12 * 4);
    float*  aNode = (float*)alloc((size_t)NA * 48 * 4);
    float*  cNode = (float*)alloc((size_t)NC * 48 * 4);
    float*  accAI = (float*)alloc((size_t)NA * 48 * 4);
    float*  cntAI = (float*)alloc((size_t)NA * 4);
    float*  accAX = (float*)alloc((size_t)NA * 48 * 4);
    float*  cntAX = (float*)alloc((size_t)NA * 4);
    float*  accCI = (float*)alloc((size_t)NC * 48 * 4);
    float*  cntCI = (float*)alloc((size_t)NC * 4);
    float*  accCX = (float*)alloc((size_t)NC * 48 * 4);
    float*  cntCX = (float*)alloc((size_t)NC * 4);
    bf16_t* convIn = (bf16_t*)alloc((size_t)CH * 64 * 2);
    float*  convW  = (float*)alloc((size_t)CH * 384 * 4);
    float*  hsum   = (float*)alloc((size_t)NC * 84 * 4);
    float*  hcnt   = (float*)alloc((size_t)NC * 4);
    bf16_t* hbuf   = (bf16_t*)alloc((size_t)NC * 96 * 2);

    // weight conversion (all small; done every call, deterministic)
    MlpP atomEdge = prep(getMlp(4), 14, 12, 12);
    MlpP cgEdge   = prep(getMlp(8), 14, 12, 12);
    MlpP crossEdge= prep(getMlp(12), 8, 12, 12);
    MlpP convAtom[3], convCg[3], convC2A[3], convA2C[3];
    for (int l = 0; l < 3; ++l) {
        convAtom[l] = prep(getMlp(16 + 4 * l), 36, 36, H_TP_NUMEL[l]);
        convCg[l]   = prep(getMlp(28 + 4 * l), 36, 36, H_TP_NUMEL[l]);
        convC2A[l]  = prep(getMlp(40 + 4 * l), 36, 36, H_TP_NUMEL[l]);
        convA2C[l]  = prep(getMlp(52 + 4 * l), 36, 36, H_TP_NUMEL[l]);
    }
    MlpP dense = prep(getMlp(64), 84, 256, 256);

    // node embeddings into zero-padded 48-wide buffers
    (void)hipMemsetAsync(aNode, 0, (size_t)NA * 48 * 4, stream);
    (void)hipMemsetAsync(cNode, 0, (size_t)NC * 48 * 4, stream);
    emb_kernel<<<g1(NA * 12), 256, 0, stream>>>(z, atom_emb, aNode, NA);
    emb_kernel<<<g1(NC * 12), 256, 0, stream>>>(cg_z, cg_emb, cNode, NC);

    // edge geometry + features
    pair_edge_kernel<<<g1(EA), 256, 0, stream>>>(xyz, z, nbr, NP_A, 12.f, aFeat, aSh, aSrc, aDst);
    pair_edge_kernel<<<g1(EC), 256, 0, stream>>>(cg_xyz, cg_z, cg_nbr, NP_C, 30.f, cFeat, cSh, cSrc, cDst);
    cross_edge_kernel<<<g1(EX), 256, 0, stream>>>(xyz, cg_xyz, mapping, NA, 30.f, xFeat, xSh, xSrc, xDst);

    runMlp(aFeat, EA, atomEdge, aEdge, 0);
    runMlp(cFeat, EC, cgEdge,   cEdge, 0);
    runMlp(xFeat, EX, crossEdge, xEdge, 0);

    for (int l = 0; l < 3; ++l) {
        int numel = H_TP_NUMEL[l], odim = H_TP_OUTDIM[l];
        (void)hipMemsetAsync(accAI, 0, (size_t)NA * 48 * 4, stream);
        (void)hipMemsetAsync(cntAI, 0, (size_t)NA * 4, stream);
        (void)hipMemsetAsync(accAX, 0, (size_t)NA * 48 * 4, stream);
        (void)hipMemsetAsync(cntAX, 0, (size_t)NA * 4, stream);
        if (l != 2) {
            (void)hipMemsetAsync(accCI, 0, (size_t)NC * 48 * 4, stream);
            (void)hipMemsetAsync(cntCI, 0, (size_t)NC * 4, stream);
            (void)hipMemsetAsync(accCX, 0, (size_t)NC * 48 * 4, stream);
            (void)hipMemsetAsync(cntCX, 0, (size_t)NC * 4, stream);
        }
        // atom intra-conv (chunked)
        for (int e0 = 0; e0 < EA; e0 += CH) {
            int n = (EA - e0 < CH) ? (EA - e0) : CH;
            gather_conv_kernel<<<g1(n * 64), 256, 0, stream>>>(aEdge, aNode, aNode, aSrc, aDst, e0, n, convIn);
            runMlp(convIn, n, convAtom[l], convW, 0);
            tp_conv_kernel<<<g1(n), 256, 0, stream>>>(aNode, aSrc, aDst, convW, aSh, e0, n, l, numel, accAI, cntAI);
        }
        // cg -> atom (cross edges): node_attr = cNode, segment over atoms
        gather_conv_kernel<<<g1(EX * 64), 256, 0, stream>>>(xEdge, aNode, cNode, xSrc, xDst, 0, EX, convIn);
        runMlp(convIn, EX, convC2A[l], convW, 0);
        tp_conv_kernel<<<g1(EX), 256, 0, stream>>>(cNode, xSrc, xDst, convW, xSh, 0, EX, l, numel, accAX, cntAX);

        if (l != 2) {
            // cg intra-conv
            for (int e0 = 0; e0 < EC; e0 += CH) {
                int n = (EC - e0 < CH) ? (EC - e0) : CH;
                gather_conv_kernel<<<g1(n * 64), 256, 0, stream>>>(cEdge, cNode, cNode, cSrc, cDst, e0, n, convIn);
                runMlp(convIn, n, convCg[l], convW, 0);
                tp_conv_kernel<<<g1(n), 256, 0, stream>>>(cNode, cSrc, cDst, convW, cSh, e0, n, l, numel, accCI, cntCI);
            }
            // atom -> cg: node_attr = aNode at x_atom, segment over mapping
            gather_conv_kernel<<<g1(EX * 64), 256, 0, stream>>>(xEdge, aNode, cNode, xSrc, xDst, 0, EX, convIn);
            runMlp(convIn, EX, convA2C[l], convW, 0);
            tp_conv_kernel<<<g1(EX), 256, 0, stream>>>(aNode, /*src=*/xDst, /*dst=*/xSrc, convW, xSh, 0, EX, l, numel, accCX, cntCX);
        }
        node_update_kernel<<<g1(NA * odim), 256, 0, stream>>>(aNode, accAI, cntAI, accAX, cntAX, NA, odim);
        if (l != 2)
            node_update_kernel<<<g1(NC * odim), 256, 0, stream>>>(cNode, accCI, cntCI, accCX, cntCX, NC, odim);
    }

    // pooling + final dense (tanh hidden) -> d_out (3000 x 256)
    (void)hipMemsetAsync(hsum, 0, (size_t)NC * 84 * 4, stream);
    (void)hipMemsetAsync(hcnt, 0, (size_t)NC * 4, stream);
    hpool_kernel<<<g1(NA), 256, 0, stream>>>(aNode, cNode, mapping, NA, hsum, hcnt);
    hdiv_kernel<<<g1(NC * 96), 256, 0, stream>>>(hsum, hcnt, hbuf, NC);
    runMlp(hbuf, NC, dense, (float*)d_out, 1);
}
